// RetentionLayer_36490042146955
// MI455X (gfx1250) — matrix operand order repaired and checked by reading, not compiled, not run on hardware
//
#include <hip/hip_runtime.h>
#include <math.h>

typedef __attribute__((ext_vector_type(16))) __bf16 v16bf;
typedef __attribute__((ext_vector_type(8)))  __bf16 v8bf;
typedef __attribute__((ext_vector_type(8)))  float  v8f;

static constexpr int Bb = 4, Ll = 2048, Dd = 1024, Hh = 4, HDs = 256;
static constexpr int Mrows = Bb * Ll; // 8192

// ------------------------------------------------------------ feature probes
#if defined(__has_builtin)
#if __has_builtin(__builtin_amdgcn_global_load_async_to_lds_b128)
#define RET_HAS_ASYNC 1
#endif
#endif
#ifndef RET_HAS_ASYNC
#define RET_HAS_ASYNC 0
#endif

#if RET_HAS_ASYNC
typedef int v4i __attribute__((vector_size(16)));
typedef __attribute__((address_space(1))) v4i as1_v4i;
typedef __attribute__((address_space(3))) v4i as3_v4i;
__device__ __forceinline__ void async_copy16(const void* gsrc, void* ldst) {
  __builtin_amdgcn_global_load_async_to_lds_b128((as1_v4i*)gsrc, (as3_v4i*)ldst, 0, 0);
}
__device__ __forceinline__ void wait_async0() {
#if __has_builtin(__builtin_amdgcn_s_wait_asynccnt)
  __builtin_amdgcn_s_wait_asynccnt(0);
#else
  asm volatile("s_wait_asynccnt 0x0" ::: "memory");
#endif
}
#endif

// ---------------------------------------------------------------- helpers
__device__ __forceinline__ v8f zero8() {
  v8f z;
#pragma unroll
  for (int i = 0; i < 8; ++i) z[i] = 0.0f;
  return z;
}

__device__ __forceinline__ v16bf cat16(v8bf lo, v8bf hi) {
  v16bf r;
#pragma unroll
  for (int i = 0; i < 8; ++i) { r[i] = lo[i]; r[i + 8] = hi[i]; }
  return r;
}

// A fragment (16x32 bf16, MxK): lanes 0-15: M=lane, K in {0..7,16..23};
// lanes 16-31: M=lane-16, K in {8..15,24..31}. Two contiguous 16B runs per lane.
__device__ __forceinline__ v16bf load_a(const __bf16* base, int ld, int row0, int k0, int lane) {
  const int half = (lane >> 4) & 1;
  const int m    = lane & 15;
  const __bf16* p = base + (size_t)(row0 + m) * ld + (k0 + half * 8);
  v8bf lo = *(const v8bf*)p;
  v8bf hi = *(const v8bf*)(p + 16);
  return cat16(lo, hi);
}

// B fragment (32x16 bf16, KxN) from a source laid out [n][k] (k contiguous):
// lane n = lane&15, K in {0..7,16..23} (lanes 0-15) / {8..15,24..31} (lanes 16-31).
__device__ __forceinline__ v16bf load_b(const __bf16* base, int ld, int n0, int k0, int lane) {
  const int half = (lane >> 4) & 1;
  const int n    = lane & 15;
  const __bf16* p = base + (size_t)(n0 + n) * ld + (k0 + half * 8);
  v8bf lo = *(const v8bf*)p;
  v8bf hi = *(const v8bf*)(p + 16);
  return cat16(lo, hi);
}

__device__ __forceinline__ v8f wmma_bf16(v16bf a, v16bf b, v8f c) {
  return __builtin_amdgcn_wmma_f32_16x16x32_bf16(false, a, false, b, (short)0, c, false, false);
}

// ---------------------------------------------------------------- fp32 -> bf16
__global__ void cvt_f32_bf16_kernel(const float* __restrict__ in, __bf16* __restrict__ out, int n) {
  int i = blockIdx.x * blockDim.x + threadIdx.x;
  const int stride = gridDim.x * blockDim.x;
  for (; i < n; i += stride) out[i] = (__bf16)in[i];
}

// ---------------------------------------------------------------- GEMM: out = A @ W^T
// Block tile 64(M) x 256(N), 8 waves as 2x4, wave tile 32x64 = 2x4 WMMA tiles.
// mode 0: bf16 store; mode 1: fp32 silu store; mode 2: fp32 store.
__global__ __launch_bounds__(256) void gemm_xw_kernel(const __bf16* __restrict__ A,
                                                      const __bf16* __restrict__ W,
                                                      void* __restrict__ out,
                                                      int mode) {
  const int tid  = threadIdx.x;
  const int lane = tid & 31;
  const int wid  = tid >> 5;
  const int wm   = wid >> 2;
  const int wn   = wid & 3;
  const int row0 = blockIdx.y * 64 + wm * 32;
  const int col0 = blockIdx.x * 256 + wn * 64;

  v8f acc[2][4];
#pragma unroll
  for (int i = 0; i < 2; ++i)
#pragma unroll
    for (int j = 0; j < 4; ++j) acc[i][j] = zero8();

#pragma unroll 4
  for (int k0 = 0; k0 < Dd; k0 += 32) {
    if (k0 + 128 < Dd)
      __builtin_prefetch(A + (size_t)(row0 + (lane & 15)) * Dd + k0 + 128, 0, 1);
    v16bf a0 = load_a(A, Dd, row0,      k0, lane);
    v16bf a1 = load_a(A, Dd, row0 + 16, k0, lane);
#pragma unroll
    for (int t = 0; t < 4; ++t) {
      v16bf b = load_b(W, Dd, col0 + t * 16, k0, lane);
      acc[0][t] = wmma_bf16(a0, b, acc[0][t]);
      acc[1][t] = wmma_bf16(a1, b, acc[1][t]);
    }
  }

  const int half = (lane >> 4) & 1;
  const int cn   = lane & 15;
#pragma unroll
  for (int mt = 0; mt < 2; ++mt)
#pragma unroll
    for (int t = 0; t < 4; ++t)
#pragma unroll
      for (int r = 0; r < 8; ++r) {
        const int row = row0 + mt * 16 + half * 8 + r;
        const int col = col0 + t * 16 + cn;
        const float v = acc[mt][t][r];
        if (mode == 0) {
          ((__bf16*)out)[(size_t)row * Dd + col] = (__bf16)v;
        } else if (mode == 1) {
          ((float*)out)[(size_t)row * Dd + col] = v / (1.0f + __expf(-v));
        } else {
          ((float*)out)[(size_t)row * Dd + col] = v;
        }
      }
}

// ---------------------------------------------------------------- retention core
// Per (b, h, 64-row i-block): O[i,:] = sum_{j: j-i>=0} (q_i . k_j) * decay^(j-i) * v_j
// (reference mask is upper-triangular), then LayerNorm over hd=256, * gate, bf16 Y.
__global__ __launch_bounds__(256) void retention_kernel(
    const __bf16* __restrict__ Qb, const __bf16* __restrict__ Kb,
    const __bf16* __restrict__ Vb, const float* __restrict__ gate,
    const float* __restrict__ gamma, const float* __restrict__ beta,
    __bf16* __restrict__ Yb) {
  __shared__ __bf16 sVt[HDs][72];   // V^T tile [d][j] (B operand of S@V)
  __shared__ __bf16 sS[64][72];     // masked S tile (A operand of S@V)
#if RET_HAS_ASYNC
  __shared__ __bf16 sK[64][264];    // K tile [j][d], async-staged (B operand of QK^T)
#endif
  __shared__ float  sRel[128];      // decay^(x-64), x in [1,127]
  __shared__ float  sSum[64], sSq[64], sMu[64], sRs[64];

  const int tid  = threadIdx.x;
  const int lane = tid & 31;
  const int wid  = tid >> 5;
  const int half = (lane >> 4) & 1;
  const int ln15 = lane & 15;

  const int b  = blockIdx.y >> 2;
  const int h  = blockIdx.y & 3;
  const int i0 = blockIdx.x * 64;
  const size_t rbase = (size_t)b * Ll;

  float d64;
  { // per-head decay, relative table, 64-step factor
    const float lmin  = __logf(1.0f / 32.0f);
    const float lmax  = __logf(1.0f / 512.0f);
    const float decay = 1.0f - __expf(lmin + (float)h * (lmax - lmin) / 3.0f);
    const float l2d   = __log2f(decay);
    d64 = exp2f(64.0f * l2d);
    for (int t = tid; t < 128; t += 256) sRel[t] = exp2f((float)(t - 64) * l2d);
  }
  if (tid < 64) { sSum[tid] = 0.0f; sSq[tid] = 0.0f; }

  const __bf16* Qh = Qb + h * HDs;
  const __bf16* Kh = Kb + h * HDs;
  const __bf16* Vh = Vb + h * HDs;

  // O-phase mapping: 2x4 waves, wave = rows wm*32..+32, cols wn*64..+64
  const int wm = wid >> 2, wn = wid & 3;
  v8f oc[2][4];
#pragma unroll
  for (int i = 0; i < 2; ++i)
#pragma unroll
    for (int j = 0; j < 4; ++j) oc[i][j] = zero8();

  // S-phase mapping: wave -> (m-tile sm, n-tiles snp..snp+1) of 64x64 S
  const int sm  = wid >> 1;
  const int snp = (wid & 1) * 2;

  // Q tile is loop-invariant: preload this wave's 8 A-fragments (K=256)
  v16bf qa[8];
#pragma unroll
  for (int kk8 = 0; kk8 < 8; ++kk8)
    qa[kk8] = load_a(Qh, Dd, (int)rbase + i0 + sm * 16, kk8 * 32, lane);

  float base = 1.0f;  // decay^(j0 - i0)
  for (int j0 = i0; j0 < Ll; j0 += 64) {
    __syncthreads();  // previous iteration's readers of sS/sVt/sK are done

    { // stage V^T into LDS (transpose 64 x 256)
      const int jr  = tid >> 2;
      const int dc0 = (tid & 3) * 64;
      const __bf16* vp = Vh + (rbase + j0 + jr) * (size_t)Dd + dc0;
#pragma unroll
      for (int it = 0; it < 8; ++it) {
        v8bf t8 = *(const v8bf*)(vp + it * 8);
#pragma unroll
        for (int e = 0; e < 8; ++e) sVt[dc0 + it * 8 + e][jr] = t8[e];
      }
    }

#if RET_HAS_ASYNC
    { // async DMA of K tile [64 x 256] straight into LDS (ASYNCcnt path)
      const int jr  = tid >> 2;
      const int dc0 = (tid & 3) * 64;
      const __bf16* kp = Kh + (rbase + j0 + jr) * (size_t)Dd + dc0;
#pragma unroll
      for (int c = 0; c < 8; ++c)
        async_copy16(kp + c * 8, &sK[jr][dc0 + c * 8]);
      wait_async0();
    }
#endif
    __syncthreads();  // staged K (and V^T) visible to all waves

    { // S = Q K^T over hd=256 (8 K-steps of 32), this wave's 2 tiles
      v8f sc0 = zero8(), sc1 = zero8();
#pragma unroll
      for (int kk8 = 0; kk8 < 8; ++kk8) {
        const int kk = kk8 * 32;
#if RET_HAS_ASYNC
        v16bf b0 = load_b(&sK[0][0], 264, snp * 16,       kk, lane);
        v16bf b1 = load_b(&sK[0][0], 264, (snp + 1) * 16, kk, lane);
#else
        v16bf b0 = load_b(Kh, Dd, (int)rbase + j0 + snp * 16,       kk, lane);
        v16bf b1 = load_b(Kh, Dd, (int)rbase + j0 + (snp + 1) * 16, kk, lane);
#endif
        sc0 = wmma_bf16(qa[kk8], b0, sc0);
        sc1 = wmma_bf16(qa[kk8], b1, sc1);
      }
#pragma unroll
      for (int t = 0; t < 2; ++t) {
        v8f sc = t ? sc1 : sc0;
        const int jl = (snp + t) * 16 + ln15;
#pragma unroll
        for (int r = 0; r < 8; ++r) {
          const int il  = sm * 16 + half * 8 + r;
          const int dif = jl - il;  // dist = (j0-i0) + dif
          const bool ok = (j0 > i0) || (dif >= 0);
          const float w = ok ? base * sRel[dif + 64] : 0.0f;
          sS[il][jl] = (__bf16)(sc[r] * w);
        }
      }
    }
    __syncthreads();  // sS visible

    // O += S @ V  (K = 64 -> two K-steps of 32)
#pragma unroll
    for (int ks = 0; ks < 64; ks += 32) {
      v16bf a0 = load_a(&sS[0][0], 72, wm * 32,      ks, lane);
      v16bf a1 = load_a(&sS[0][0], 72, wm * 32 + 16, ks, lane);
#pragma unroll
      for (int t = 0; t < 4; ++t) {
        v16bf bb = load_b(&sVt[0][0], 72, wn * 64 + t * 16, ks, lane);
        oc[0][t] = wmma_bf16(a0, bb, oc[0][t]);
        oc[1][t] = wmma_bf16(a1, bb, oc[1][t]);
      }
    }
    base *= d64;
  }
  __syncthreads();

  // ---- LayerNorm over hd=256 (rows split across 4 wn-waves -> LDS reduce) ----
#pragma unroll
  for (int mt = 0; mt < 2; ++mt)
#pragma unroll
    for (int r = 0; r < 8; ++r) {
      float s = 0.0f, q = 0.0f;
#pragma unroll
      for (int t = 0; t < 4; ++t) { const float v = oc[mt][t][r]; s += v; q += v * v; }
#pragma unroll
      for (int off = 1; off < 16; off <<= 1) {
        s += __shfl_xor(s, off, 32);
        q += __shfl_xor(q, off, 32);
      }
      if (ln15 == 0) {
        const int il = wm * 32 + mt * 16 + half * 8 + r;
        atomicAdd(&sSum[il], s);
        atomicAdd(&sSq[il],  q);
      }
    }
  __syncthreads();
  if (tid < 64) {
    const float mu  = sSum[tid] * (1.0f / 256.0f);
    const float var = sSq[tid] * (1.0f / 256.0f) - mu * mu;
    sMu[tid] = mu;
    sRs[tid] = rsqrtf(var + 1e-5f);
  }
  __syncthreads();

  // Y = LN(O) * gate (gate already silu'd), bf16
#pragma unroll
  for (int mt = 0; mt < 2; ++mt)
#pragma unroll
    for (int t = 0; t < 4; ++t)
#pragma unroll
      for (int r = 0; r < 8; ++r) {
        const int il = wm * 32 + mt * 16 + half * 8 + r;
        const int d  = wn * 64 + t * 16 + ln15;
        const float v = (oc[mt][t][r] - sMu[il]) * sRs[il] * gamma[d] + beta[d];
        const size_t row = rbase + i0 + il;
        const float g = gate[row * Dd + h * HDs + d];
        Yb[row * Dd + h * HDs + d] = (__bf16)(v * g);
      }
}

// ---------------------------------------------------------------- launch
extern "C" void kernel_launch(void* const* d_in, const int* in_sizes, int n_in,
                              void* d_out, int out_size, void* d_ws, size_t ws_size,
                              hipStream_t stream) {
  (void)in_sizes; (void)n_in; (void)out_size; (void)ws_size;
  const float* x     = (const float*)d_in[0];
  const float* wq    = (const float*)d_in[1];
  const float* wk    = (const float*)d_in[2];
  const float* wv    = (const float*)d_in[3];
  const float* wg    = (const float*)d_in[4];
  const float* wo    = (const float*)d_in[5];
  const float* gamma = (const float*)d_in[6];
  const float* beta  = (const float*)d_in[7];

  char* ws = (char*)d_ws;
  auto bump = [&](size_t bytes) -> void* {
    void* p = (void*)ws;
    ws += (bytes + 255) & ~(size_t)255;
    return p;
  };
  __bf16* xb   = (__bf16*)bump((size_t)Mrows * Dd * 2);
  __bf16* wqb  = (__bf16*)bump((size_t)Dd * Dd * 2);
  __bf16* wkb  = (__bf16*)bump((size_t)Dd * Dd * 2);
  __bf16* wvb  = (__bf16*)bump((size_t)Dd * Dd * 2);
  __bf16* wgb  = (__bf16*)bump((size_t)Dd * Dd * 2);
  __bf16* wob  = (__bf16*)bump((size_t)Dd * Dd * 2);
  __bf16* Qb   = (__bf16*)bump((size_t)Mrows * Dd * 2);
  __bf16* Kb   = (__bf16*)bump((size_t)Mrows * Dd * 2);
  __bf16* Vb   = (__bf16*)bump((size_t)Mrows * Dd * 2);
  float*  gate = (float*) bump((size_t)Mrows * Dd * 4);
  __bf16* Yb   = (__bf16*)bump((size_t)Mrows * Dd * 2);

  dim3 blk(256);
  cvt_f32_bf16_kernel<<<dim3(2048), blk, 0, stream>>>(x,  xb,  Mrows * Dd);
  cvt_f32_bf16_kernel<<<dim3(512),  blk, 0, stream>>>(wq, wqb, Dd * Dd);
  cvt_f32_bf16_kernel<<<dim3(512),  blk, 0, stream>>>(wk, wkb, Dd * Dd);
  cvt_f32_bf16_kernel<<<dim3(512),  blk, 0, stream>>>(wv, wvb, Dd * Dd);
  cvt_f32_bf16_kernel<<<dim3(512),  blk, 0, stream>>>(wg, wgb, Dd * Dd);
  cvt_f32_bf16_kernel<<<dim3(512),  blk, 0, stream>>>(wo, wob, Dd * Dd);

  dim3 ggrid(Dd / 256, Mrows / 64);
  gemm_xw_kernel<<<ggrid, blk, 0, stream>>>(xb, wqb, Qb,   0);
  gemm_xw_kernel<<<ggrid, blk, 0, stream>>>(xb, wkb, Kb,   0);
  gemm_xw_kernel<<<ggrid, blk, 0, stream>>>(xb, wvb, Vb,   0);
  gemm_xw_kernel<<<ggrid, blk, 0, stream>>>(xb, wgb, gate, 1);

  retention_kernel<<<dim3(Ll / 64, Bb * Hh), blk, 0, stream>>>(Qb, Kb, Vb, gate, gamma, beta, Yb);

  gemm_xw_kernel<<<ggrid, blk, 0, stream>>>(Yb, wob, d_out, 2);
}
